// enhanced_39453569581175
// MI455X (gfx1250) — matrix operand order repaired and checked by reading, not compiled, not run on hardware
//
#include <hip/hip_runtime.h>

#define D     128
#define NS    8192
#define NQ    16384
#define TOPK  32

typedef __attribute__((ext_vector_type(16))) _Float16 v16h;
typedef __attribute__((ext_vector_type(8)))  _Float16 v8h;
typedef __attribute__((ext_vector_type(4)))  _Float16 v4h;
typedef __attribute__((ext_vector_type(8)))  float    v8f;

// ---------------- workspace layout (bytes) ----------------
#define XS_BYTES   (NS * D * 2)                 // normalized s_emb as f16
#define PART_OFF_B (XS_BYTES)                   // 64x128 proto partial sums
#define PART_BYTES (64 * D * 4)
#define VV_OFF_B   (PART_OFF_B + PART_BYTES)    // V[2][128]
#define VV_BYTES   (2 * D * 4)
#define UU_OFF_B   (VV_OFF_B + VV_BYTES)        // u[2][128] = Wq^T K_j
#define UU_BYTES   (2 * D * 4)
#define CC_OFF_B   (UU_OFF_B + UU_BYTES)        // c[2]   = bq . K_j

// ---------------- LDS layout for the fused sim/topk kernel ----------------
#define KSTRIDE    8196                          // 8192 + 4 u16 pad (bank spread, even)
#define KEYS_BYTES (16 * KSTRIDE * 2)            // 262272 B of 16-bit sort keys
#define SMEM_BYTES (KEYS_BYTES + 16*32*4 /*red*/ + 16*TOPK*4 /*sel*/ + 32*4 /*cnt*/)

// f32 -> f16 -> order-preserving u16 key, fully branchless:
//   negative: b ^ 0xFFFF (= ~b)   positive: b ^ 0x8000
__device__ __forceinline__ unsigned short f2key(float v) {
  _Float16 h = (_Float16)v;
  unsigned short b;
  __builtin_memcpy(&b, &h, 2);
  const unsigned short m = (unsigned short)(((short)b) >> 15);   // 0xFFFF or 0x0000
  return (unsigned short)(b ^ (m | (unsigned short)0x8000u));
}

// ---------------- Kernel 1: row-normalize s_emb into f16 ----------------
__global__ void normalize_kernel(const float* __restrict__ s, _Float16* __restrict__ xs) {
  const int wave = threadIdx.x >> 5, lane = threadIdx.x & 31;
  const int row  = blockIdx.x * 8 + wave;
  const float4 v = *((const float4*)(s + (size_t)row * D) + lane);
  __shared__ float red[256];
  red[threadIdx.x] = v.x*v.x + v.y*v.y + v.z*v.z + v.w*v.w;
  __syncthreads();
  float tot = 0.f;
#pragma unroll
  for (int l = 0; l < 32; ++l) tot += red[wave * 32 + l];
  const float inv = rsqrtf(tot);
  v4h o;
  o[0] = (_Float16)(v.x * inv); o[1] = (_Float16)(v.y * inv);
  o[2] = (_Float16)(v.z * inv); o[3] = (_Float16)(v.w * inv);
  *(v4h*)(xs + (size_t)row * D + lane * 4) = o;
}

// ---------------- Kernel 2a: proto partial sums (64 blocks x 128 rows) ----
__global__ void proto_partial_kernel(const float* __restrict__ s, float* __restrict__ part) {
  const int t = threadIdx.x, b = blockIdx.x;
  const float* p = s + (size_t)b * 128 * D + t;
  float acc = 0.f;
  for (int r = 0; r < 128; ++r) acc += p[(size_t)r * D];
  part[b * D + t] = acc;
}

// ---------------- Kernel 2b: protos + folded attention precompute ---------
// u_j = Wq^T K_j ; c_j = bq . K_j ; V_j = Wv proto_j + bv
__global__ void attn_precompute_kernel(const float* __restrict__ part,
                                       const float* __restrict__ Wq, const float* __restrict__ bq,
                                       const float* __restrict__ Wk, const float* __restrict__ bk,
                                       const float* __restrict__ Wv, const float* __restrict__ bv,
                                       float* __restrict__ Uw, float* __restrict__ Cw,
                                       float* __restrict__ Vw) {
  const int t = threadIdx.x;
  __shared__ float proto[2][D];
  __shared__ float Ksh[2][D];
  float neg = 0.f, pos = 0.f;
  for (int b = 0;  b < 32; ++b) neg += part[b * D + t];
  for (int b = 32; b < 64; ++b) pos += part[b * D + t];
  proto[0][t] = pos * (1.0f / 4096.0f);   // proto[0] = pos (second half)
  proto[1][t] = neg * (1.0f / 4096.0f);   // proto[1] = neg (first half)
  __syncthreads();
  float k0 = bk[t], k1 = bk[t], v0 = bv[t], v1 = bv[t];
  for (int i = 0; i < D; ++i) {
    const float wk = Wk[t * D + i], wv = Wv[t * D + i];
    k0 += wk * proto[0][i];  k1 += wk * proto[1][i];
    v0 += wv * proto[0][i];  v1 += wv * proto[1][i];
  }
  Ksh[0][t] = k0; Ksh[1][t] = k1;
  Vw[t] = v0; Vw[D + t] = v1;
  __syncthreads();
  float u0 = 0.f, u1 = 0.f;
  for (int i = 0; i < D; ++i) {
    const float w = Wq[i * D + t];        // Wq^T
    u0 += w * Ksh[0][i];  u1 += w * Ksh[1][i];
  }
  Uw[t] = u0; Uw[D + t] = u1;
  if (t < 2) {
    float c = 0.f;
    for (int i = 0; i < D; ++i) c += bq[i] * Ksh[t][i];
    Cw[t] = c;
  }
}

// ---------------- Kernel 3: fused WMMA sim -> top-32 -> aggregate ---------
__global__ __launch_bounds__(512) void sim_topk_agg_kernel(
    const _Float16* __restrict__ xs, const float* __restrict__ s,
    const float* __restrict__ alpha_p, float* __restrict__ out_s) {
  extern __shared__ unsigned char smem[];
  unsigned short* keys = (unsigned short*)smem;
  int* red = (int*)(smem + KEYS_BYTES);
  int* sel = red + 16 * 32;
  int* cnt = sel + 16 * TOPK;

  const int wave = threadIdx.x >> 5;
  const int lane = threadIdx.x & 31;
  const int half = lane >> 4;
  const int mr   = lane & 15;
  const int rowBlock = blockIdx.x;

  // ---- Phase 1: 16x8192 sim stripe via v_wmma_f32_16x16x32_f16 ----
  // A (16x32 f16) per-lane layout: lanes 0-15 row=lane, K={0..7,16..23};
  //                                lanes 16-31 row=lane-16, K={8..15,24..31}
  v16h a[4];
  {
    const _Float16* arow = xs + (size_t)(rowBlock * 16 + mr) * D;
#pragma unroll
    for (int ch = 0; ch < 4; ++ch) {
      const int kb = ch * 32 + half * 8;
      v8h lo = *(const v8h*)(arow + kb);
      v8h hi = *(const v8h*)(arow + kb + 16);
      a[ch] = __builtin_shufflevector(lo, hi, 0,1,2,3,4,5,6,7,8,9,10,11,12,13,14,15);
    }
  }
  for (int t = wave; t < NS / 16; t += 16) {
    const int col0 = t * 16;
    // B (32x16 f16): lane column = mr; lanes 0-15 K={0..7,16..23}, lanes 16-31 K={8..15,24..31}
    const _Float16* brow = xs + (size_t)(col0 + mr) * D;
    v8f c = {};
#pragma unroll
    for (int ch = 0; ch < 4; ++ch) {
      const int kb = ch * 32 + half * 8;
      v8h blo = *(const v8h*)(brow + kb);
      v8h bhi = *(const v8h*)(brow + kb + 16);
      v16h b = __builtin_shufflevector(blo, bhi, 0,1,2,3,4,5,6,7,8,9,10,11,12,13,14,15);
      c = __builtin_amdgcn_wmma_f32_16x16x32_f16(false, a[ch], false, b,
                                                 (short)0, c, false, false);
    }
    // C layout: VGPR j -> (row = j + 8*half, col = col0 + mr); store as sort key
    const int n = col0 + mr;
#pragma unroll
    for (int j = 0; j < 8; ++j)
      keys[(j + 8 * half) * KSTRIDE + n] = f2key(c[j]);
  }
  __syncthreads();

  // ---- Phase 2: per-row (wave `wave`) exact top-32 via 16-step binary search ----
  // Keys are scanned as packed u32 (two keys per ds load).
  const unsigned int* krow32 = (const unsigned int*)(keys + wave * KSTRIDE);
  int lo = 0, hi = 65535;
  for (int it = 0; it < 16; ++it) {
    const int mid = (lo + hi + 1) >> 1;
    const unsigned int m32 = (unsigned int)mid;
    int local = 0;
    for (int j = lane; j < NS / 2; j += 32) {
      const unsigned int p = krow32[j];
      local += ((p & 0xFFFFu) >= m32) ? 1 : 0;
      local += ((p >> 16)     >= m32) ? 1 : 0;
    }
    red[wave * 32 + lane] = local;
    __syncthreads();
    int tot = 0;
#pragma unroll
    for (int l = 0; l < 32; ++l) tot += red[wave * 32 + l];
    __syncthreads();
    if (tot >= TOPK) lo = mid; else hi = mid - 1;
  }
  const unsigned int T = (unsigned int)lo;   // max t with count(>=t) >= 32
  if (lane == 0) { cnt[wave * 2] = 0; cnt[wave * 2 + 1] = 0; }
  __syncthreads();
  for (int j = lane; j < NS / 2; j += 32) {
    const unsigned int p = krow32[j];
    const unsigned int klo = p & 0xFFFFu, khi = p >> 16;
    if (klo > T) {
      const int slot = atomicAdd(&cnt[wave * 2], 1);
      if (slot < TOPK) sel[wave * TOPK + slot] = 2 * j;
    }
    if (khi > T) {
      const int slot = atomicAdd(&cnt[wave * 2], 1);
      if (slot < TOPK) sel[wave * TOPK + slot] = 2 * j + 1;
    }
  }
  __syncthreads();
  const int c1 = cnt[wave * 2];
  for (int j = lane; j < NS / 2; j += 32) {
    const unsigned int p = krow32[j];
    const unsigned int klo = p & 0xFFFFu, khi = p >> 16;
    if (klo == T) {
      const int slot = atomicAdd(&cnt[wave * 2 + 1], 1);
      if (c1 + slot < TOPK) sel[wave * TOPK + c1 + slot] = 2 * j;
    }
    if (khi == T) {
      const int slot = atomicAdd(&cnt[wave * 2 + 1], 1);
      if (c1 + slot < TOPK) sel[wave * TOPK + c1 + slot] = 2 * j + 1;
    }
  }
  __syncthreads();

  // ---- Phase 3: gather-sum selected rows, write enhanced_s ----
  const float alpha = *alpha_p;
  const int grow = rowBlock * 16 + wave;
  float ax = 0.f, ay = 0.f, az = 0.f, aw = 0.f;
#pragma unroll 4
  for (int k = 0; k < TOPK; ++k) {
    const int idx = sel[wave * TOPK + k];
    const float4 v = *((const float4*)(s + (size_t)idx * D) + lane);
    ax += v.x; ay += v.y; az += v.z; aw += v.w;
  }
  const float4 b = *((const float4*)(s + (size_t)grow * D) + lane);
  float4 o;
  o.x = b.x + alpha * ax; o.y = b.y + alpha * ay;
  o.z = b.z + alpha * az; o.w = b.w + alpha * aw;
  *((float4*)(out_s + (size_t)grow * D) + lane) = o;
}

// ---------------- Kernel 4: folded streaming query attention --------------
__global__ void query_attn_kernel(const float* __restrict__ q,
                                  const float* __restrict__ U, const float* __restrict__ Cw,
                                  const float* __restrict__ V, const float* __restrict__ alpha_p,
                                  float* __restrict__ out_q) {
  const int wave = threadIdx.x >> 5, lane = threadIdx.x & 31;
  const int row = blockIdx.x * 8 + wave;
  const float4 qv = *((const float4*)(q + (size_t)row * D) + lane);
  const float4 u0 = *((const float4*)U + lane);
  const float4 u1 = *((const float4*)(U + D) + lane);
  __shared__ float red[2][256];
  red[0][threadIdx.x] = qv.x*u0.x + qv.y*u0.y + qv.z*u0.z + qv.w*u0.w;
  red[1][threadIdx.x] = qv.x*u1.x + qv.y*u1.y + qv.z*u1.z + qv.w*u1.w;
  __syncthreads();
  float l0 = 0.f, l1 = 0.f;
#pragma unroll
  for (int l = 0; l < 32; ++l) { l0 += red[0][wave*32 + l]; l1 += red[1][wave*32 + l]; }
  const float invS = 0.0883883476483184f;   // 1/sqrt(128)
  l0 = (l0 + Cw[0]) * invS;
  l1 = (l1 + Cw[1]) * invS;
  const float m  = fmaxf(l0, l1);
  const float e0 = __expf(l0 - m), e1 = __expf(l1 - m);
  const float a0 = e0 / (e0 + e1), a1 = 1.0f - a0;
  const float al = *alpha_p;
  const float4 v0 = *((const float4*)V + lane);
  const float4 v1 = *((const float4*)(V + D) + lane);
  float4 o;
  o.x = qv.x + al * (a0 * v0.x + a1 * v1.x);
  o.y = qv.y + al * (a0 * v0.y + a1 * v1.y);
  o.z = qv.z + al * (a0 * v0.z + a1 * v1.z);
  o.w = qv.w + al * (a0 * v0.w + a1 * v1.w);
  *((float4*)(out_q + (size_t)row * D) + lane) = o;
}

// ---------------- launcher ----------------
extern "C" void kernel_launch(void* const* d_in, const int* in_sizes, int n_in,
                              void* d_out, int out_size, void* d_ws, size_t ws_size,
                              hipStream_t stream) {
  const float* s_emb      = (const float*)d_in[0];
  const float* q_emb      = (const float*)d_in[1];
  const float* Wq         = (const float*)d_in[2];
  const float* bq         = (const float*)d_in[3];
  const float* Wk         = (const float*)d_in[4];
  const float* bk         = (const float*)d_in[5];
  const float* Wv         = (const float*)d_in[6];
  const float* bv         = (const float*)d_in[7];
  const float* alpha_msg  = (const float*)d_in[8];
  const float* alpha_attn = (const float*)d_in[9];

  float* out_s = (float*)d_out;
  float* out_q = out_s + (size_t)NS * D;

  unsigned char* ws = (unsigned char*)d_ws;
  _Float16* xs  = (_Float16*)ws;
  float* part   = (float*)(ws + PART_OFF_B);
  float* Vw     = (float*)(ws + VV_OFF_B);
  float* Uw     = (float*)(ws + UU_OFF_B);
  float* Cw     = (float*)(ws + CC_OFF_B);

  normalize_kernel<<<NS / 8, 256, 0, stream>>>(s_emb, xs);
  proto_partial_kernel<<<64, 128, 0, stream>>>(s_emb, part);
  attn_precompute_kernel<<<1, 128, 0, stream>>>(part, Wq, bq, Wk, bk, Wv, bv, Uw, Cw, Vw);

  (void)hipFuncSetAttribute((const void*)sim_topk_agg_kernel,
                            hipFuncAttributeMaxDynamicSharedMemorySize, SMEM_BYTES);
  sim_topk_agg_kernel<<<NS / 16, 512, SMEM_BYTES, stream>>>(xs, s_emb, alpha_msg, out_s);

  query_attn_kernel<<<NQ / 8, 256, 0, stream>>>(q_emb, Uw, Cw, Vw, alpha_attn, out_q);
}
